// CondConvMobileNetV2_87359634801432
// MI455X (gfx1250) — matrix operand order repaired and checked by reading, not compiled, not run on hardware
//
#include <hip/hip_runtime.h>
#include <hip/hip_bf16.h>
#include <math.h>

typedef __attribute__((ext_vector_type(16))) _Float16 v16h;
typedef __attribute__((ext_vector_type(8)))  float    v8f;

#define BATCH 32

__device__ __forceinline__ unsigned cc_pkf16(float lo, float hi) {
  union { _Float16 h[2]; unsigned u; } t;
  t.h[0] = (_Float16)lo;
  t.h[1] = (_Float16)hi;
  return t.u;
}

__device__ __forceinline__ int cc_imin(int a, int b) { return a < b ? a : b; }

// ------------------------------------------------------------------
// Fused 1x1-conv GEMM via WMMA, LDS-staged, fast/edge tile split.
//   out[b,m,n] = BN( sum_k W[m,k]*in[b,k,n] )  (+resid) (ReLU6?)
// Block = 128 threads (4 waves). Tile: 32 rows (M) x 64 cols (N).
// Interior tiles: float4 global loads + packed-pair LDS stores.
// Edge tiles: clamped-index loads (always in-bounds) + select-zero,
// so even the edge path is branch-free (no exec-mask churn).
// Each wave: one B fragment, two M-half WMMAs per K-step of 32.
// ------------------------------------------------------------------
__global__ __launch_bounds__(128) void cc_gemm1x1_wmma(
    const float* __restrict__ in,   // [B, K, N]
    const float* __restrict__ w,    // [M, K]
    const float* __restrict__ bng, const float* __restrict__ bnb,
    const float* __restrict__ bnm, const float* __restrict__ bnv,
    const float* __restrict__ resid,  // [B, M, N] or nullptr
    float* __restrict__ out,          // [B, M, N]
    int M, int N, int K, int MTiles, int NTiles, int relu6, int use_res)
{
  // K-pair packed f16 tiles, padded stride 17 dwords (17 coprime 64 banks)
  __shared__ unsigned As[32][17];   // [m][kpair]
  __shared__ unsigned Bs[64][17];   // [n][kpair]
  _Float16* AsH = (_Float16*)As;    // half view: index m*34 + k
  _Float16* BsH = (_Float16*)Bs;    // half view: index n*34 + k

  int tile = blockIdx.x;
  int nt = tile % NTiles; tile /= NTiles;
  int mt = tile % MTiles; int b = tile / MTiles;
  int mBase = mt * 32;
  int nBase = nt * 64;
  const float* inB = in + (size_t)b * K * N;

  int tid  = threadIdx.x;
  int wave = tid >> 5;
  int lane = tid & 31;
  int lm = lane & 15;
  int hi = lane >> 4;                // 0: lanes 0-15, 1: lanes 16-31

  bool mFull = (mBase + 32 <= M);
  bool nFull = (nBase + 64 <= N) && ((N & 3) == 0);

  v8f acc0 = {};
  v8f acc1 = {};

  for (int kk = 0; kk < K; kk += 32) {
    bool kFull = (kk + 32 <= K);

    // ---- A tile: 32 rows x 32 k ----
    if (mFull && kFull) {
      // fast: each thread converts 8 consecutive k of one row
      int m  = tid >> 2;
      int kq = (tid & 3) * 8;
      const float* src = w + (size_t)(mBase + m) * K + kk + kq;
      float4 p0 = *(const float4*)(src);
      float4 p1 = *(const float4*)(src + 4);
      int kp = kq >> 1;
      As[m][kp + 0] = cc_pkf16(p0.x, p0.y);
      As[m][kp + 1] = cc_pkf16(p0.z, p0.w);
      As[m][kp + 2] = cc_pkf16(p1.x, p1.y);
      As[m][kp + 3] = cc_pkf16(p1.z, p1.w);
    } else {
      // edge: clamped (always in-bounds) load + select-zero, branch-free
#pragma unroll
      for (int e4 = 0; e4 < 8; ++e4) {
        int e = tid + e4 * 128;      // 0..1023
        int m = e >> 5, k = e & 31;
        bool ok = (mBase + m < M) && (kk + k < K);
        int mi = cc_imin(mBase + m, M - 1);
        int ki = cc_imin(kk + k, K - 1);
        float v = w[(size_t)mi * K + ki];
        AsH[m * 34 + k] = (_Float16)(ok ? v : 0.f);
      }
    }

    // ---- B tile: 32 k-rows x 64 cols ----
    if (nFull && kFull) {
      // fast: each thread loads two adjacent k-rows x 8 cols, packs k-pairs
      int kp = tid >> 3;             // 0..15 (k pair index)
      int nq = (tid & 7) * 8;        // 0..56
      const float* r0 = inB + (size_t)(kk + 2 * kp) * N + nBase + nq;
      const float* r1 = r0 + N;
      float4 a0 = *(const float4*)(r0);
      float4 a1 = *(const float4*)(r0 + 4);
      float4 b0 = *(const float4*)(r1);
      float4 b1 = *(const float4*)(r1 + 4);
      Bs[nq + 0][kp] = cc_pkf16(a0.x, b0.x);
      Bs[nq + 1][kp] = cc_pkf16(a0.y, b0.y);
      Bs[nq + 2][kp] = cc_pkf16(a0.z, b0.z);
      Bs[nq + 3][kp] = cc_pkf16(a0.w, b0.w);
      Bs[nq + 4][kp] = cc_pkf16(a1.x, b1.x);
      Bs[nq + 5][kp] = cc_pkf16(a1.y, b1.y);
      Bs[nq + 6][kp] = cc_pkf16(a1.z, b1.z);
      Bs[nq + 7][kp] = cc_pkf16(a1.w, b1.w);
    } else {
      // edge: clamped load + select-zero, branch-free
#pragma unroll
      for (int e4 = 0; e4 < 16; ++e4) {
        int e = tid + e4 * 128;      // 0..2047
        int k = e >> 6, n = e & 63;
        bool ok = (nBase + n < N) && (kk + k < K);
        int ni = cc_imin(nBase + n, N - 1);
        int ki = cc_imin(kk + k, K - 1);
        float v = inB[(size_t)ki * N + ni];
        BsH[n * 34 + k] = (_Float16)(ok ? v : 0.f);
      }
    }
    __syncthreads();

    // ---- build fragments from LDS (dword loads, conflict-free) ----
    v16h bfrag{};
    {
      const unsigned* brow = &Bs[wave * 16 + lm][0];
#pragma unroll
      for (int j = 0; j < 8; ++j) {
        union { unsigned u; _Float16 h[2]; } t;
        t.u = brow[hi * 8 + j];      // B: lanes 0-15 K=0..15, lanes 16-31 K=16..31
        bfrag[2 * j]     = t.h[0];
        bfrag[2 * j + 1] = t.h[1];
      }
    }
    v16h a0{};
    v16h a1{};
    {
      const unsigned* ar0 = &As[lm][0];
      const unsigned* ar1 = &As[16 + lm][0];
#pragma unroll
      for (int j = 0; j < 8; ++j) {
        int kp = hi * 8 + j;  // A: lanes 0-15 K=0..15, lanes 16-31 K=16..31
        union { unsigned u; _Float16 h[2]; } t0, t1;
        t0.u = ar0[kp];
        t1.u = ar1[kp];
        a0[2 * j]     = t0.h[0];
        a0[2 * j + 1] = t0.h[1];
        a1[2 * j]     = t1.h[0];
        a1[2 * j + 1] = t1.h[1];
      }
    }
    acc0 = __builtin_amdgcn_wmma_f32_16x16x32_f16(false, a0, false, bfrag,
                                                  (short)0, acc0, false, false);
    acc1 = __builtin_amdgcn_wmma_f32_16x16x32_f16(false, a1, false, bfrag,
                                                  (short)0, acc1, false, false);
    __syncthreads();
  }

  // ---- epilogue: BN (+resid) (+ReLU6), C/D layout: VGPR r -> row r+8*hi ----
  int n = nBase + wave * 16 + lm;
  if (n < N) {
#pragma unroll
    for (int r = 0; r < 8; ++r) {
      int mr = mBase + r + hi * 8;
      if (mr < M) {
        float s    = bng[mr] * rsqrtf(bnv[mr] + 1e-5f);
        float bias = bnb[mr] - bnm[mr] * s;
        float y = acc0[r] * s + bias;
        size_t o = ((size_t)b * M + mr) * N + n;
        if (use_res) y += resid[o];
        if (relu6)   y = fminf(fmaxf(y, 0.f), 6.f);
        out[o] = y;
      }
    }
#pragma unroll
    for (int r = 0; r < 8; ++r) {
      int mr = mBase + 16 + r + hi * 8;
      if (mr < M) {
        float s    = bng[mr] * rsqrtf(bnv[mr] + 1e-5f);
        float bias = bnb[mr] - bnm[mr] * s;
        float y = acc1[r] * s + bias;
        size_t o = ((size_t)b * M + mr) * N + n;
        if (use_res) y += resid[o];
        if (relu6)   y = fminf(fmaxf(y, 0.f), 6.f);
        out[o] = y;
      }
    }
  }
}

// ------------------------------------------------------------------
// Stem: 3x3 stride-2 conv (3 -> 32) + BN + ReLU6. One thread / output elem.
// ------------------------------------------------------------------
__global__ void cc_stem(const float* __restrict__ x,  // [B,3,224,224]
                        const float* __restrict__ w,  // [32,3,3,3]
                        const float* __restrict__ g, const float* __restrict__ bb,
                        const float* __restrict__ mm, const float* __restrict__ vv,
                        float* __restrict__ out)      // [B,32,112,112]
{
  int idx = blockIdx.x * blockDim.x + threadIdx.x;
  const int total = BATCH * 32 * 112 * 112;
  if (idx >= total) return;
  int xo = idx % 112; int t = idx / 112;
  int yo = t % 112;  t /= 112;
  int co = t % 32;   int b = t / 32;
  float acc = 0.f;
#pragma unroll
  for (int ci = 0; ci < 3; ++ci)
#pragma unroll
    for (int ky = 0; ky < 3; ++ky)
#pragma unroll
      for (int kx = 0; kx < 3; ++kx) {
        int iy = 2 * yo - 1 + ky;
        int ix = 2 * xo - 1 + kx;
        if (iy >= 0 && iy < 224 && ix >= 0 && ix < 224)
          acc += x[(((size_t)b * 3 + ci) * 224 + iy) * 224 + ix] *
                 w[((co * 3 + ci) * 3 + ky) * 3 + kx];
      }
  float s = g[co] * rsqrtf(vv[co] + 1e-5f);
  float y = acc * s + (bb[co] - mm[co] * s);
  out[idx] = fminf(fmaxf(y, 0.f), 6.f);
}

// ------------------------------------------------------------------
// Global average pool over HW: one 256-thread block per (b,c).
// ------------------------------------------------------------------
__global__ void cc_gap(const float* __restrict__ in, float* __restrict__ out, int HW)
{
  __shared__ float red[256];
  int bc = blockIdx.x;
  const float* p = in + (size_t)bc * HW;
  float s = 0.f;
  for (int i = threadIdx.x; i < HW; i += 256) s += p[i];
  red[threadIdx.x] = s;
  __syncthreads();
  for (int off = 128; off > 0; off >>= 1) {
    if ((int)threadIdx.x < off) red[threadIdx.x] += red[threadIdx.x + off];
    __syncthreads();
  }
  if (threadIdx.x == 0) out[bc] = red[0] / (float)HW;
}

// ------------------------------------------------------------------
// Routing: r[b,e] = sigmoid( pool[b,:] . route_w[e,:] + route_b[e] )
// ------------------------------------------------------------------
__global__ void cc_route(const float* __restrict__ pool, const float* __restrict__ rw,
                         const float* __restrict__ rb, float* __restrict__ r, int C)
{
  int idx = blockIdx.x * blockDim.x + threadIdx.x;
  if (idx >= BATCH * 4) return;
  int b = idx >> 2, e = idx & 3;
  const float* p = pool + (size_t)b * C;
  const float* w = rw + (size_t)e * C;
  float z = rb[e];
  for (int c = 0; c < C; ++c) z += p[c] * w[c];
  r[idx] = 1.f / (1.f + expf(-z));
}

// ------------------------------------------------------------------
// Expert mixing: cw[b,c,k] = sum_e r[b,e] * cc_w[e,c,k]   (k = 0..8)
// ------------------------------------------------------------------
__global__ void cc_mix(const float* __restrict__ r, const float* __restrict__ ccw,
                       float* __restrict__ cw, int C)
{
  int idx = blockIdx.x * blockDim.x + threadIdx.x;
  int total = BATCH * C * 9;
  if (idx >= total) return;
  int k = idx % 9; int t = idx / 9;
  int c = t % C;   int b = t / C;
  float acc = 0.f;
#pragma unroll
  for (int e = 0; e < 4; ++e)
    acc += r[b * 4 + e] * ccw[((size_t)e * C + c) * 9 + k];
  cw[idx] = acc;
}

// ------------------------------------------------------------------
// Per-sample conditional depthwise 3x3 (pad 1) + BN + ReLU6.
// ------------------------------------------------------------------
__global__ void cc_dw(const float* __restrict__ in,  // [B,C,Hin,Hin]
                      const float* __restrict__ cw,  // [B,C,9]
                      const float* __restrict__ g, const float* __restrict__ bb,
                      const float* __restrict__ mm, const float* __restrict__ vv,
                      float* __restrict__ out,       // [B,C,Hout,Hout]
                      int C, int Hin, int Hout, int stride)
{
  int idx = blockIdx.x * blockDim.x + threadIdx.x;
  int total = BATCH * C * Hout * Hout;
  if (idx >= total) return;
  int x = idx % Hout; int t = idx / Hout;
  int y = t % Hout;  t /= Hout;
  int c = t % C;     int b = t / C;
  const float* pin = in + ((size_t)b * C + c) * Hin * Hin;
  const float* kw  = cw + ((size_t)b * C + c) * 9;
  float acc = 0.f;
#pragma unroll
  for (int ky = 0; ky < 3; ++ky)
#pragma unroll
    for (int kx = 0; kx < 3; ++kx) {
      int iy = y * stride - 1 + ky;
      int ix = x * stride - 1 + kx;
      if (iy >= 0 && iy < Hin && ix >= 0 && ix < Hin)
        acc += pin[iy * Hin + ix] * kw[ky * 3 + kx];
    }
  float s = g[c] * rsqrtf(vv[c] + 1e-5f);
  float yv = acc * s + (bb[c] - mm[c] * s);
  out[idx] = fminf(fmaxf(yv, 0.f), 6.f);
}

// ------------------------------------------------------------------
// Final FC: out[b,n] = g[b,:] . fc_w[n,:] + fc_b[n]
// ------------------------------------------------------------------
__global__ void cc_fc(const float* __restrict__ gfeat, const float* __restrict__ fw,
                      const float* __restrict__ fb, float* __restrict__ out)
{
  int idx = blockIdx.x * blockDim.x + threadIdx.x;
  if (idx >= BATCH * 1000) return;
  int n = idx % 1000, b = idx / 1000;
  const float* p = gfeat + (size_t)b * 1280;
  const float* w = fw + (size_t)n * 1280;
  float acc = fb[n];
  for (int c = 0; c < 1280; ++c) acc += p[c] * w[c];
  out[idx] = acc;
}

// ------------------------------------------------------------------
// Host orchestration
// ------------------------------------------------------------------
struct BlkCfg { int inp, oup, stride, t, hin; };
static const BlkCfg BL[17] = {
  {32, 16, 1, 1, 112},
  {16, 24, 2, 6, 112},
  {24, 24, 1, 6, 56},
  {24, 32, 2, 6, 56},
  {32, 32, 1, 6, 28},
  {32, 32, 1, 6, 28},
  {32, 64, 2, 6, 28},
  {64, 64, 1, 6, 14},
  {64, 64, 1, 6, 14},
  {64, 64, 1, 6, 14},
  {64, 96, 1, 6, 14},
  {96, 96, 1, 6, 14},
  {96, 96, 1, 6, 14},
  {96, 160, 2, 6, 14},
  {160, 160, 1, 6, 7},
  {160, 160, 1, 6, 7},
  {160, 320, 1, 6, 7},
};

extern "C" void kernel_launch(void* const* d_in, const int* in_sizes, int n_in,
                              void* d_out, int out_size, void* d_ws, size_t ws_size,
                              hipStream_t stream) {
  (void)in_sizes; (void)n_in; (void)out_size; (void)ws_size;
  int ii = 0;
  auto F = [&](void) -> const float* { return (const float*)d_in[ii++]; };

  float* ws = (float*)d_ws;
  // workspace layout (elements)
  const size_t O0 = 0;            // current/output buffer (<= 12.85M)
  const size_t O1 = 13000000;     // expand buffer        (<= 38.6M)
  const size_t O2 = 51600000;     // depthwise buffer     (<= 14.5M)
  const size_t OPOOL = 66100000;  // pooled feats [32,<=1280]
  const size_t ORGATE = 66145000; // routing gates [32,4]
  const size_t OCW = 66146000;    // mixed dw kernels [32,<=960,9]

  auto launch_gemm = [&](const float* in, const float* w,
                         const float* g, const float* b,
                         const float* m, const float* v,
                         const float* res, float* out,
                         int M, int N, int K, int relu6, int use_res) {
    int MTiles = (M + 31) / 32, NTiles = (N + 63) / 64;
    dim3 grid((unsigned)(BATCH * MTiles * NTiles));
    cc_gemm1x1_wmma<<<grid, dim3(128), 0, stream>>>(
        in, w, g, b, m, v, res, out, M, N, K, MTiles, NTiles, relu6, use_res);
  };

  // ---- inputs in setup_inputs() dict order ----
  const float* x = F();
  const float* stem_w = F();
  const float *sg = F(), *sb = F(), *sm = F(), *sv = F();

  // stem conv + BN + ReLU6 -> ws[O0] : [32,32,112,112]
  {
    int total = BATCH * 32 * 112 * 112;
    cc_stem<<<dim3((total + 255) / 256), dim3(256), 0, stream>>>(
        x, stem_w, sg, sb, sm, sv, ws + O0);
  }

  float* cur = ws + O0;
  for (int bi = 0; bi < 17; ++bi) {
    const BlkCfg& c = BL[bi];
    int hid = c.inp * c.t;
    int hin = c.hin, hout = c.hin / c.stride;
    int Nin = hin * hin, Nout = hout * hout;

    const float *exp_w = nullptr, *eg = nullptr, *eb = nullptr, *em = nullptr, *ev = nullptr;
    if (c.t != 1) { exp_w = F(); eg = F(); eb = F(); em = F(); ev = F(); }
    const float* route_w = F();
    const float* route_b = F();
    const float* ccw = F();
    const float *dg = F(), *db = F(), *dm = F(), *dv = F();
    const float* pw_w = F();
    const float *pg = F(), *pb = F(), *pm = F(), *pv = F();

    // expand 1x1 (WMMA GEMM) + BN + ReLU6
    float* hbuf = cur;
    if (c.t != 1) {
      hbuf = ws + O1;
      launch_gemm(cur, exp_w, eg, eb, em, ev, nullptr, hbuf,
                  hid, Nin, c.inp, /*relu6=*/1, /*res=*/0);
    }

    // routing: GAP -> linear -> sigmoid
    cc_gap<<<dim3((unsigned)(BATCH * hid)), dim3(256), 0, stream>>>(hbuf, ws + OPOOL, Nin);
    cc_route<<<dim3((BATCH * 4 + 63) / 64), dim3(64), 0, stream>>>(
        ws + OPOOL, route_w, route_b, ws + ORGATE, hid);

    // per-sample expert kernel mixing
    {
      int total = BATCH * hid * 9;
      cc_mix<<<dim3((total + 255) / 256), dim3(256), 0, stream>>>(
          ws + ORGATE, ccw, ws + OCW, hid);
    }

    // conditional depthwise 3x3 + BN + ReLU6 -> ws[O2]
    {
      int total = BATCH * hid * Nout;
      cc_dw<<<dim3((total + 255) / 256), dim3(256), 0, stream>>>(
          hbuf, ws + OCW, dg, db, dm, dv, ws + O2, hid, hin, hout, c.stride);
    }

    // pointwise 1x1 (WMMA GEMM) + BN (+ residual), write back to O0
    int use_res = (c.stride == 1 && c.inp == c.oup) ? 1 : 0;
    launch_gemm(ws + O2, pw_w, pg, pb, pm, pv, use_res ? cur : nullptr, ws + O0,
                c.oup, Nout, hid, /*relu6=*/0, use_res);
    cur = ws + O0;
  }

  // head 1x1 conv 320->1280 + BN + ReLU6 (WMMA GEMM)
  const float* head_w = F();
  const float *hg = F(), *hb = F(), *hm = F(), *hv = F();
  const float* fc_w = F();
  const float* fc_b = F();

  launch_gemm(cur, head_w, hg, hb, hm, hv, nullptr, ws + O1,
              1280, 49, 320, /*relu6=*/1, /*res=*/0);

  // global average pool -> [32,1280]
  cc_gap<<<dim3((unsigned)(BATCH * 1280)), dim3(256), 0, stream>>>(ws + O1, ws + OPOOL, 49);

  // classifier
  cc_fc<<<dim3((BATCH * 1000 + 255) / 256), dim3(256), 0, stream>>>(
      ws + OPOOL, fc_w, fc_b, (float*)d_out);
}
